// InterpretableMultiHeadAttention_36575941493113
// MI455X (gfx1250) — compile-verified, matrix operand order checked by reading, not run on hardware
//
#include <hip/hip_runtime.h>
#include <hip/hip_bf16.h>

typedef __attribute__((ext_vector_type(16))) _Float16 v16h;
typedef __attribute__((ext_vector_type(8)))  _Float16 v8h;
typedef __attribute__((ext_vector_type(4)))  _Float16 v4h;
typedef __attribute__((ext_vector_type(8)))  float    v8f;
typedef __attribute__((ext_vector_type(4)))  float    v4f;

#define D_MODEL 1024
#define N_HEADS 16
#define D_K     64
#define SEQ_T   2048
#define BATCH   2

// ---------- WMMA helper (CDNA5 16x16x32 f16 -> f32 accum) ----------
__device__ __forceinline__ v8f wmma_f16(v16h a, v16h b, v8f c) {
  // (neg_a, A, neg_b, B, c_mod, C, reuse_a, reuse_b)
  return __builtin_amdgcn_wmma_f32_16x16x32_f16(false, a, false, b, (short)0, c,
                                                false, false);
}

// Convert 8 contiguous fp32 to 8 f16
__device__ __forceinline__ v8h cvt8f32(const float* p) {
  v4f a = *(const v4f*)p;
  v4f b = *(const v4f*)(p + 4);
  v8h r;
#pragma unroll
  for (int i = 0; i < 4; ++i) { r[i] = (_Float16)a[i]; r[4 + i] = (_Float16)b[i]; }
  return r;
}

// A/B fragment per CDNA5 16-bit layout: lane (b0 = lane/16) holds
// K = k0 + 8*b0 + {0..7} in halves [0..7], K = k0 + 16 + 8*b0 + {0..7} in [8..15].
__device__ __forceinline__ v16h frag_f32(const float* row, int k0, int b0) {
  v8h lo = cvt8f32(row + k0 + 8 * b0);
  v8h hi = cvt8f32(row + k0 + 16 + 8 * b0);
  v16h f;
#pragma unroll
  for (int i = 0; i < 8; ++i) { f[i] = lo[i]; f[8 + i] = hi[i]; }
  return f;
}

__device__ __forceinline__ v16h frag_f16(const _Float16* row, int k0, int b0) {
  v8h lo = *(const v8h*)(row + k0 + 8 * b0);
  v8h hi = *(const v8h*)(row + k0 + 16 + 8 * b0);
  v16h f;
#pragma unroll
  for (int i = 0; i < 8; ++i) { f[i] = lo[i]; f[8 + i] = hi[i]; }
  return f;
}

// ---------- elementwise fp32 -> f16 (one pass; L2/HBM cheap) ----------
__global__ __launch_bounds__(256) void cvt_f32_f16(const float* __restrict__ x,
                                                   _Float16* __restrict__ y,
                                                   int n4) {
  int i = blockIdx.x * 256 + threadIdx.x;
  if (i < n4) {
    v4f a = ((const v4f*)x)[i];
    v4h r;
#pragma unroll
    for (int j = 0; j < 4; ++j) r[j] = (_Float16)a[j];
    ((v4h*)y)[i] = r;
  }
}

// ---------- GEMM: Y = X @ W^T + bias, all-f16 fragments ----------
// One 16x32 output tile per wave32 (A fragment reused across 2 WMMAs).
template <bool OUT_F16>
__global__ __launch_bounds__(128) void gemm_f16_wmma(
    const _Float16* __restrict__ X, const _Float16* __restrict__ W,
    const float* __restrict__ bias, void* __restrict__ Yv,
    int M, int N, int K) {
  const int lane = threadIdx.x & 31;
  const int wave = threadIdx.x >> 5;
  const int Nt2 = N >> 5;                       // 32-wide column tiles
  const int tile = blockIdx.x * 4 + wave;
  const int mt = tile / Nt2;
  const int nt2 = tile % Nt2;
  const int rsel = lane & 15;
  const int b0 = lane >> 4;

  const _Float16* xrow = X + (size_t)(mt * 16 + rsel) * K;
  const _Float16* wrow0 = W + (size_t)(nt2 * 32 + rsel) * K;
  const _Float16* wrow1 = W + (size_t)(nt2 * 32 + 16 + rsel) * K;

  v8f acc0 = {}, acc1 = {};
#pragma unroll 2
  for (int k0 = 0; k0 < K; k0 += 32) {
    if (k0 + 64 < K) {  // hint next K-step into cache (global_prefetch_b8)
      __builtin_prefetch(xrow + k0 + 64, 0, 1);
      __builtin_prefetch(wrow0 + k0 + 64, 0, 1);
      __builtin_prefetch(wrow1 + k0 + 64, 0, 1);
    }
    v16h a = frag_f16(xrow, k0, b0);
    v16h bm0 = frag_f16(wrow0, k0, b0);   // B(k,n) = W[n][k]
    v16h bm1 = frag_f16(wrow1, k0, b0);
    acc0 = wmma_f16(a, bm0, acc0);
    acc1 = wmma_f16(a, bm1, acc1);
  }

  const int col0 = nt2 * 32 + rsel;       // D: N = lane%16
  const int col1 = col0 + 16;
  const float bb0 = bias[col0];
  const float bb1 = bias[col1];
  const int mbase = mt * 16 + 8 * b0;     // D: M = v + 8*(lane/16)
  if (OUT_F16) {
    _Float16* Y = (_Float16*)Yv;
#pragma unroll
    for (int v = 0; v < 8; ++v) {
      Y[(size_t)(mbase + v) * N + col0] = (_Float16)(acc0[v] + bb0);
      Y[(size_t)(mbase + v) * N + col1] = (_Float16)(acc1[v] + bb1);
    }
  } else {
    float* Y = (float*)Yv;
#pragma unroll
    for (int v = 0; v < 8; ++v) {
      Y[(size_t)(mbase + v) * N + col0] = acc0[v] + bb0;
      Y[(size_t)(mbase + v) * N + col1] = acc1[v] + bb1;
    }
  }
}

// ---------- Fused attention: scores -> softmax -> attn write -> P@V ----------
// Block = 128 threads (4 waves) per (b, h, 16-query tile).
// LDS: 16x2048 fp32 score stripe (128KB) resident through all phases ->
// the 512MB attention matrix crosses HBM exactly once.
__global__ __launch_bounds__(128) void attn_fused(
    const _Float16* __restrict__ Qh, const _Float16* __restrict__ Kh,
    const _Float16* __restrict__ Vh, float* __restrict__ attn,
    _Float16* __restrict__ ctx) {
  extern __shared__ __attribute__((aligned(16))) char smem[];
  float* S = (float*)smem;                          // [16][2048] fp32
  float* red = (float*)(smem + 16 * SEQ_T * 4);     // [128] partials
  float* rowstat = red + 128;                       // [0:16) max, [16:32) 1/sum
  _Float16* vt = (_Float16*)(rowstat + 32);         // [4 waves][16*32] f16

  const int tid = threadIdx.x;
  const int lane = tid & 31;
  const int wave = tid >> 5;
  const int b0 = lane >> 4;
  const int rsel = lane & 15;

  const int qt = blockIdx.x & (SEQ_T / 16 - 1);     // 128 q-tiles
  const int bh = blockIdx.x >> 7;
  const int bat = bh >> 4;
  const int h = bh & 15;

  // Q fragments for this 16-row tile (d_k = 64 -> two K=32 chunks)
  const size_t rowQ = (size_t)(bat * SEQ_T + qt * 16 + rsel) * D_MODEL + h * D_K;
  const v16h aq0 = frag_f16(Qh + rowQ, 0, b0);
  const v16h aq1 = frag_f16(Qh + rowQ, 32, b0);

  // ---- Phase 1: S = (Q K^T)/sqrt(d_k); waves stride over key tiles ----
  for (int kt = wave; kt < SEQ_T / 16; kt += 4) {
    const size_t rowK = (size_t)(bat * SEQ_T + kt * 16 + rsel) * D_MODEL + h * D_K;
    v16h bk0 = frag_f16(Kh + rowK, 0, b0);          // B(k,n)=K[key n][k]
    v16h bk1 = frag_f16(Kh + rowK, 32, b0);
    v8f acc = {};
    acc = wmma_f16(aq0, bk0, acc);
    acc = wmma_f16(aq1, bk1, acc);
#pragma unroll
    for (int v = 0; v < 8; ++v)
      S[(size_t)(v + 8 * b0) * SEQ_T + kt * 16 + rsel] = acc[v] * 0.125f;
  }
  __syncthreads();

  // ---- Phase 2: row softmax in LDS, vectorized v4f (b128 DS ops) ----
  const int r = tid & 15;   // row
  const int c = tid >> 4;   // 8 chunks/row, 16B-interleaved
  v4f* Srow4 = (v4f*)(S + (size_t)r * SEQ_T);
  v4f vmx = {-3.402823466e38f, -3.402823466e38f, -3.402823466e38f,
             -3.402823466e38f};
  for (int i = 0; i < SEQ_T / 32; ++i) {
    v4f t = Srow4[c + 8 * i];
#pragma unroll
    for (int j = 0; j < 4; ++j) vmx[j] = fmaxf(vmx[j], t[j]);
  }
  red[c * 16 + r] = fmaxf(fmaxf(vmx[0], vmx[1]), fmaxf(vmx[2], vmx[3]));
  __syncthreads();
  if (c == 0) {
    float mm = red[r];
#pragma unroll
    for (int j = 1; j < 8; ++j) mm = fmaxf(mm, red[j * 16 + r]);
    rowstat[r] = mm;
  }
  __syncthreads();
  const float rmax = rowstat[r];
  v4f vsum = {0.f, 0.f, 0.f, 0.f};
  for (int i = 0; i < SEQ_T / 32; ++i) {
    v4f t = Srow4[c + 8 * i];
#pragma unroll
    for (int j = 0; j < 4; ++j) t[j] = __expf(t[j] - rmax);
    Srow4[c + 8 * i] = t;   // keep UNnormalized exp resident
    vsum += t;
  }
  red[c * 16 + r] = vsum[0] + vsum[1] + vsum[2] + vsum[3];
  __syncthreads();
  if (c == 0) {
    float ss = 0.f;
#pragma unroll
    for (int j = 0; j < 8; ++j) ss += red[j * 16 + r];
    rowstat[16 + r] = 1.0f / ss;
  }
  __syncthreads();

  // ---- Phase 3: single coalesced HBM write of normalized attn weights ----
  {
    float* base = attn + ((size_t)bh * SEQ_T + qt * 16) * SEQ_T;
    for (int mrow = 0; mrow < 16; ++mrow) {
      const float inv = rowstat[16 + mrow];
      for (int idx = tid * 4; idx < SEQ_T; idx += 128 * 4) {
        v4f vsc = *(v4f*)&S[(size_t)mrow * SEQ_T + idx];
        vsc *= inv;
        *(v4f*)(base + (size_t)mrow * SEQ_T + idx) = vsc;
      }
    }
  }

  // ---- Phase 4: context = P @ V; each wave owns one 16-wide d slice ----
  _Float16* vtw = vt + wave * (16 * 32);            // private V^T pane [d][key]
  const int d0 = wave * 16;
  v8f acc = {};
  for (int kc = 0; kc < SEQ_T / 32; ++kc) {
    // stage V[key, d0..d0+15] transposed into LDS (one key per lane)
    const _Float16* vr =
        Vh + (size_t)(bat * SEQ_T + kc * 32 + lane) * D_MODEL + h * D_K + d0;
    v8h v0 = *(const v8h*)vr;
    v8h v1 = *(const v8h*)(vr + 8);
#pragma unroll
    for (int j = 0; j < 8; ++j) {
      vtw[j * 32 + lane] = v0[j];
      vtw[(8 + j) * 32 + lane] = v1[j];
    }
    __syncthreads();  // uniform; orders LDS stores before fragment loads
    v16h ap = frag_f32(S + (size_t)rsel * SEQ_T, kc * 32, b0);  // P (unnorm)
    v16h bv = frag_f16(vtw + rsel * 32, 0, b0);                 // V^T pane
    acc = wmma_f16(ap, bv, acc);
    __syncthreads();
  }
  // fold 1/sum into context, store f16 at [b*T + q, h*64 + d]
#pragma unroll
  for (int v = 0; v < 8; ++v) {
    const int mrow = v + 8 * b0;
    float val = acc[v] * rowstat[16 + mrow];
    ctx[(size_t)(bat * SEQ_T + qt * 16 + mrow) * D_MODEL + h * D_K + d0 + rsel] =
        (_Float16)val;
  }
}

// ------------------------------- launcher ------------------------------------
extern "C" void kernel_launch(void* const* d_in, const int* in_sizes, int n_in,
                              void* d_out, int out_size, void* d_ws,
                              size_t ws_size, hipStream_t stream) {
  const float* q  = (const float*)d_in[0];
  const float* k  = (const float*)d_in[1];
  const float* v  = (const float*)d_in[2];
  const float* Wq = (const float*)d_in[3];
  const float* bq = (const float*)d_in[4];
  const float* Wk = (const float*)d_in[5];
  const float* bk = (const float*)d_in[6];
  const float* Wv = (const float*)d_in[7];
  const float* bv = (const float*)d_in[8];
  const float* Wo = (const float*)d_in[9];
  const float* bo = (const float*)d_in[10];

  float* out = (float*)d_out;
  float* attn = out + (size_t)BATCH * SEQ_T * D_MODEL;  // tuple: output, attn

  const size_t eAct = (size_t)BATCH * SEQ_T * D_MODEL;  // 4M elems
  const size_t eW = (size_t)D_MODEL * D_MODEL;          // 1M elems
  _Float16* Qh = (_Float16*)d_ws;      // 8MB
  _Float16* Kh = Qh + eAct;            // 8MB
  _Float16* Vh = Kh + eAct;            // 8MB
  _Float16* Ch = Vh + eAct;            // 8MB
  _Float16* Xh = Ch + eAct;            // 8MB staging (reused q/k/v)
  _Float16* Wh = Xh + eAct;            // 2MB staging (reused per weight)

  const int M = BATCH * SEQ_T, N = D_MODEL, K = D_MODEL;
  dim3 blk(128);
  dim3 gGemm((unsigned)((M / 16) * (N / 32) / 4));      // 2048 blocks
  dim3 cvtA((unsigned)(eAct / 4 / 256)), cvtW((unsigned)(eW / 4 / 256));
  dim3 cblk(256);

  // Q projection
  cvt_f32_f16<<<cvtA, cblk, 0, stream>>>(q, Xh, (int)(eAct / 4));
  cvt_f32_f16<<<cvtW, cblk, 0, stream>>>(Wq, Wh, (int)(eW / 4));
  gemm_f16_wmma<true><<<gGemm, blk, 0, stream>>>(Xh, Wh, bq, Qh, M, N, K);
  // K projection
  cvt_f32_f16<<<cvtA, cblk, 0, stream>>>(k, Xh, (int)(eAct / 4));
  cvt_f32_f16<<<cvtW, cblk, 0, stream>>>(Wk, Wh, (int)(eW / 4));
  gemm_f16_wmma<true><<<gGemm, blk, 0, stream>>>(Xh, Wh, bk, Kh, M, N, K);
  // V projection
  cvt_f32_f16<<<cvtA, cblk, 0, stream>>>(v, Xh, (int)(eAct / 4));
  cvt_f32_f16<<<cvtW, cblk, 0, stream>>>(Wv, Wh, (int)(eW / 4));
  gemm_f16_wmma<true><<<gGemm, blk, 0, stream>>>(Xh, Wh, bv, Vh, M, N, K);

  // Fused attention (136KB dynamic LDS per block; 320KB/WGP budget)
  const size_t shmem =
      (size_t)16 * SEQ_T * 4 + (128 + 32) * 4 + 4 * 16 * 32 * 2;
  dim3 gattn((unsigned)(BATCH * N_HEADS * (SEQ_T / 16)));
  attn_fused<<<gattn, blk, shmem, stream>>>(Qh, Kh, Vh, attn, Ch);

  // Output projection (fp32 out)
  cvt_f32_f16<<<cvtW, cblk, 0, stream>>>(Wo, Wh, (int)(eW / 4));
  gemm_f16_wmma<false><<<gGemm, blk, 0, stream>>>(Ch, Wh, bo, out, M, N, K);
}
